// TrustModule_76836964926320
// MI455X (gfx1250) — matrix operand order repaired and checked by reading, not compiled, not run on hardware
//
#include <hip/hip_runtime.h>
#include <cmath>

typedef _Float16 half_t;
typedef __attribute__((ext_vector_type(8)))  _Float16 v8h;
typedef __attribute__((ext_vector_type(16))) _Float16 v16h;
typedef __attribute__((ext_vector_type(8)))  float    v8f;

#define E_TOT   50000
#define D_DIM   512
#define NHEAD   8
#define HD_DIM  64
#define WWIN    6
#define ETILE   16
#define RWIN    (ETILE*WWIN)   /* 96 evo rows per block */
#define SSTR    520            /* padded LDS row stride in halves (512+8) */
#define KT16    16             /* K-tiles of 32: 512/32 */
#define HMSTR   264            /* hmid stride (256+8) */

/* ---- LDS carve (bytes) ---- */
#define OFF_EVO 0
#define SZ_EVO  (RWIN*SSTR*2)          /* 99840 */
#define OFF_KV  (OFF_EVO + SZ_EVO)
#define SZ_KV   (RWIN*SSTR*2)          /* 99840 */
#define OFF_H   (OFF_KV + SZ_KV)
#define SZ_H    (ETILE*SSTR*2)         /* 16640 : h, later pooled */
#define OFF_Q   (OFF_H + SZ_H)
#define SZ_Q    (ETILE*SSTR*2)         /* 16640 : Q, later hmid  */
#define OFF_SC  (OFF_Q + SZ_Q)
#define SZ_SC   (ETILE*NHEAD*WWIN*4)   /* 3072  : scores/attn    */
#define SMEM_BYTES (OFF_SC + SZ_SC)    /* 236032 */

/* -------- weight pre-pack: fp32 row-major [K=512][N] -> f16 WMMA-B layout --------
 * out[ ((nt*KT16 + kt)*32 + lane)*16 + j ] = W[k][n]
 *   k = kt*32 + (lane>>4)*16 + j ,  n = nt*16 + (lane&15)
 * i.e. lanes 0-15 carry K=k0..k0+15 for N=lane, lanes 16-31 carry K=k0+16..k0+31. */
__global__ __launch_bounds__(256) void pack_b_kernel(const float* __restrict__ W,
                                                     half_t* __restrict__ out, int N) {
  int idx = blockIdx.x * 256 + threadIdx.x;          // one output half per thread
  int j = idx & 15;
  int t = idx >> 4;
  int lane = t & 31;  t >>= 5;
  int kt = t % KT16;
  int nt = t / KT16;
  int k = kt * 32 + (lane >> 4) * 16 + j;
  int n = nt * 16 + (lane & 15);
  out[idx] = (half_t)W[k * N + n];
}

/* -------- register-tiled WMMA GEMM fragment: C[MF*16, NF*16] += A(LDS f16) x B(packed f16) ---- */
template <int MF, int NF>
__device__ __forceinline__ void gemm_frag(const half_t* sA, int strideA, int row0,
                                          const half_t* __restrict__ Bpk, int nt0,
                                          v8f (&acc)[MF][NF]) {
  const int lane = threadIdx.x & 31;
  const int mr = lane & 15;
  const int g  = lane >> 4;
  for (int kt = 0; kt < KT16; ++kt) {
    v16h Af[MF];
#pragma unroll
    for (int i = 0; i < MF; ++i) {
      const half_t* ap = sA + (size_t)(row0 + i * 16 + mr) * strideA + kt * 32 + g * 16;
      v8h a0 = *(const v8h*)(ap);        // K = k0 + g*16 + 0..7
      v8h a1 = *(const v8h*)(ap + 8);    // K = k0 + g*16 + 8..15
      Af[i] = __builtin_shufflevector(a0, a1, 0,1,2,3,4,5,6,7,8,9,10,11,12,13,14,15);
    }
#pragma unroll
    for (int j = 0; j < NF; ++j) {
      const v16h Bf = *(const v16h*)(Bpk + (size_t)(((nt0 + j) * KT16 + kt) * 32 + lane) * 16);
#pragma unroll
      for (int i = 0; i < MF; ++i)
        acc[i][j] = __builtin_amdgcn_wmma_f32_16x16x32_f16(
            false, Af[i], false, Bf, (short)0, acc[i][j], false, false);
    }
  }
}

/* -------- fused trust-module kernel: one block = 16 edges -------- */
__global__ __launch_bounds__(256) void trust_kernel(
    const float* __restrict__ h_i,  const float* __restrict__ evo,
    const float* __restrict__ var_i,const float* __restrict__ var_j,
    const float* __restrict__ bq,   const float* __restrict__ bk,
    const float* __restrict__ bv,   const float* __restrict__ logdecay,
    const float* __restrict__ b1,   const float* __restrict__ W2,
    const float* __restrict__ b2,
    const half_t* __restrict__ wq_pk, const half_t* __restrict__ wk_pk,
    const half_t* __restrict__ wv_pk, const half_t* __restrict__ w1_pk,
    float* __restrict__ out) {
  extern __shared__ char smem[];
  half_t* sEvo = (half_t*)(smem + OFF_EVO);   // [96][520] f16 (A for K/V GEMM)
  half_t* sKV  = (half_t*)(smem + OFF_KV);    // K tile, then V tile
  half_t* sH   = (half_t*)(smem + OFF_H);     // h tile, then pooled
  half_t* sQ   = (half_t*)(smem + OFF_Q);     // Q tile (pre-scaled), then hmid
  float*  sSc  = (float*)(smem + OFF_SC);     // [16][8][6] scores -> attn

  const int tid  = threadIdx.x;
  const int wid  = tid >> 5;
  const int lane = tid & 31;
  const int e0   = blockIdx.x * ETILE;

  /* ---- phase 0: stage h & evo tiles as f16 in LDS (float4 global loads) ---- */
  {
    const float4* hv = (const float4*)(h_i + (size_t)e0 * D_DIM);
    for (int i = tid; i < ETILE * 128; i += 256) {
      float4 v = hv[i];
      int r = i >> 7, c = (i & 127) * 4;
      half_t* d = sH + (size_t)r * SSTR + c;
      d[0] = (half_t)v.x; d[1] = (half_t)v.y; d[2] = (half_t)v.z; d[3] = (half_t)v.w;
    }
    const float4* ev = (const float4*)(evo + (size_t)e0 * WWIN * D_DIM);
    for (int i = tid; i < RWIN * 128; i += 256) {
      float4 v = ev[i];
      int r = i >> 7, c = (i & 127) * 4;
      half_t* d = sEvo + (size_t)r * SSTR + c;
      d[0] = (half_t)v.x; d[1] = (half_t)v.y; d[2] = (half_t)v.z; d[3] = (half_t)v.w;
    }
  }
  __syncthreads();

  /* ---- phase 1a: Q = h @ Wq + bq, pre-scaled by HD^-0.5, stored f16 ---- */
  {
    v8f acc[1][4] = {};
    gemm_frag<1, 4>(sH, SSTR, 0, wq_pk, wid * 4, acc);
    const int n = lane & 15, hi = lane >> 4;
#pragma unroll
    for (int j = 0; j < 4; ++j) {
      int col = (wid * 4 + j) * 16 + n;
      float bb = bq[col];
#pragma unroll
      for (int v = 0; v < 8; ++v)
        sQ[(size_t)(v + 8 * hi) * SSTR + col] = (half_t)((acc[0][j][v] + bb) * 0.125f);
    }
  }
  /* ---- phase 1b: K = evo @ Wk + bk (96x512), 3Mx4N register tiles ---- */
  for (int job = wid; job < 16; job += 8) {
    int mb = job >> 3, nb = job & 7;
    v8f acc[3][4] = {};
    gemm_frag<3, 4>(sEvo, SSTR, mb * 48, wk_pk, nb * 4, acc);
    const int n = lane & 15, hi = lane >> 4;
#pragma unroll
    for (int j = 0; j < 4; ++j) {
      int col = (nb * 4 + j) * 16 + n;
      float bb = bk[col];
#pragma unroll
      for (int i = 0; i < 3; ++i)
#pragma unroll
        for (int v = 0; v < 8; ++v)
          sKV[(size_t)(mb * 48 + i * 16 + v + 8 * hi) * SSTR + col] = (half_t)(acc[i][j][v] + bb);
    }
  }
  __syncthreads();

  /* ---- phase 2: scores[e][h][w] = (Q_scaled . K) ---- */
  for (int t = tid; t < ETILE * NHEAD * WWIN; t += 256) {
    int w = t % WWIN; int eh = t / WWIN; int h = eh & 7; int e = eh >> 3;
    const half_t* q = sQ + (size_t)e * SSTR + h * HD_DIM;
    const half_t* k = sKV + (size_t)(e * WWIN + w) * SSTR + h * HD_DIM;
    float s = 0.f;
#pragma unroll
    for (int c = 0; c < 8; ++c) {
      v8h qa = *(const v8h*)(q + c * 8);
      v8h ka = *(const v8h*)(k + c * 8);
#pragma unroll
      for (int d = 0; d < 8; ++d) s += (float)qa[d] * (float)ka[d];
    }
    sSc[t] = s;
  }
  __syncthreads();

  /* ---- phase 3: decay bias + softmax over W=6, in place ---- */
  if (tid < ETILE * NHEAD) {
    float dec = expf(logdecay[0]);
    float bias[WWIN]; float bsum = 0.f;
#pragma unroll
    for (int w = 0; w < WWIN; ++w) { bias[w] = expf(dec * (float)w); bsum += bias[w]; }
    float binv = 1.f / bsum;
    float* sc = sSc + tid * WWIN;
    float buf[WWIN]; float m = -1e30f;
#pragma unroll
    for (int w = 0; w < WWIN; ++w) { buf[w] = sc[w] + bias[w] * binv; m = fmaxf(m, buf[w]); }
    float sum = 0.f;
#pragma unroll
    for (int w = 0; w < WWIN; ++w) { buf[w] = expf(buf[w] - m); sum += buf[w]; }
    float inv = 1.f / sum;
#pragma unroll
    for (int w = 0; w < WWIN; ++w) sc[w] = buf[w] * inv;
  }
  __syncthreads();

  /* ---- phase 4: V = evo @ Wv + bv  (overwrites K buffer) ---- */
  for (int job = wid; job < 16; job += 8) {
    int mb = job >> 3, nb = job & 7;
    v8f acc[3][4] = {};
    gemm_frag<3, 4>(sEvo, SSTR, mb * 48, wv_pk, nb * 4, acc);
    const int n = lane & 15, hi = lane >> 4;
#pragma unroll
    for (int j = 0; j < 4; ++j) {
      int col = (nb * 4 + j) * 16 + n;
      float bb = bv[col];
#pragma unroll
      for (int i = 0; i < 3; ++i)
#pragma unroll
        for (int v = 0; v < 8; ++v)
          sKV[(size_t)(mb * 48 + i * 16 + v + 8 * hi) * SSTR + col] = (half_t)(acc[i][j][v] + bb);
    }
  }
  __syncthreads();

  /* ---- phase 5: pooled[e][d] = sum_w attn[e][h(d)][w] * V[e][w][d] -> sH (f16) ---- */
  for (int t = tid; t < ETILE * 128; t += 256) {
    int d4 = (t & 127) * 4, e = t >> 7;
    int h = d4 >> 6;
    const float* a = sSc + (size_t)(e * NHEAD + h) * WWIN;
    float p0 = 0, p1 = 0, p2 = 0, p3 = 0;
#pragma unroll
    for (int w = 0; w < WWIN; ++w) {
      const half_t* vp = sKV + (size_t)(e * WWIN + w) * SSTR + d4;
      float aw = a[w];
      p0 += aw * (float)vp[0]; p1 += aw * (float)vp[1];
      p2 += aw * (float)vp[2]; p3 += aw * (float)vp[3];
    }
    half_t* dst = sH + (size_t)e * SSTR + d4;
    dst[0] = (half_t)p0; dst[1] = (half_t)p1; dst[2] = (half_t)p2; dst[3] = (half_t)p3;
  }
  __syncthreads();

  /* ---- phase 6: hmid = GELU_exact(pooled @ W1 + b1) -> sQ (f16, stride 264) ---- */
  {
    v8f acc[1][2] = {};
    gemm_frag<1, 2>(sH, SSTR, 0, w1_pk, wid * 2, acc);
    const int n = lane & 15, hi = lane >> 4;
#pragma unroll
    for (int j = 0; j < 2; ++j) {
      int col = (wid * 2 + j) * 16 + n;
      float bb = b1[col];
#pragma unroll
      for (int v = 0; v < 8; ++v) {
        float x = acc[0][j][v] + bb;
        float gel = 0.5f * x * (1.0f + erff(x * 0.70710678118654752f));
        sQ[(size_t)(v + 8 * hi) * HMSTR + col] = (half_t)gel;
      }
    }
  }
  __syncthreads();

  /* ---- phase 7: logit = hmid . W2 + b2 ; confidence epilogue ; sigmoid ---- */
  for (int e = wid * 2; e < wid * 2 + 2; ++e) {
    float part = 0.f;
#pragma unroll
    for (int r = 0; r < 8; ++r) {
      int c = r * 32 + lane;
      part += (float)sQ[(size_t)e * HMSTR + c] * W2[c];
    }
#pragma unroll
    for (int off = 16; off; off >>= 1) part += __shfl_xor(part, off);

    float si = 0.f, sj = 0.f;
    const float* vi = var_i + (size_t)(e0 + e) * D_DIM;
    const float* vj = var_j + (size_t)(e0 + e) * D_DIM;
#pragma unroll
    for (int r = 0; r < 16; ++r) { si += vi[r * 32 + lane]; sj += vj[r * 32 + lane]; }
#pragma unroll
    for (int off = 16; off; off >>= 1) { si += __shfl_xor(si, off); sj += __shfl_xor(sj, off); }

    if (lane == 0) {
      float logit = part + b2[0];
      float mi = si * (1.f / 512.f), mj = sj * (1.f / 512.f);
      float ci = 1.f / (1.f + fmaxf(sqrtf(mi), 1e-6f));
      float cj = 1.f / (1.f + fmaxf(sqrtf(mj), 1e-6f));
      float lc = fmaxf(logf(sqrtf(ci * cj)), -5.0f);
      float z = logit + lc;
      out[e0 + e] = 1.f / (1.f + expf(-z));
    }
  }
}

extern "C" void kernel_launch(void* const* d_in, const int* in_sizes, int n_in,
                              void* d_out, int out_size, void* d_ws, size_t ws_size,
                              hipStream_t stream) {
  (void)in_sizes; (void)n_in; (void)out_size; (void)ws_size;
  const float* h_i   = (const float*)d_in[0];
  const float* evo   = (const float*)d_in[1];
  const float* var_i = (const float*)d_in[2];
  const float* var_j = (const float*)d_in[3];
  const float* Wq    = (const float*)d_in[4];
  const float* bq    = (const float*)d_in[5];
  const float* Wk    = (const float*)d_in[6];
  const float* bk    = (const float*)d_in[7];
  const float* Wv    = (const float*)d_in[8];
  const float* bv    = (const float*)d_in[9];
  const float* logd  = (const float*)d_in[10];
  const float* W1    = (const float*)d_in[11];
  const float* b1    = (const float*)d_in[12];
  const float* W2    = (const float*)d_in[13];
  const float* b2    = (const float*)d_in[14];
  float* out = (float*)d_out;

  half_t* ws    = (half_t*)d_ws;
  half_t* wq_pk = ws;
  half_t* wk_pk = ws + 512 * 512;
  half_t* wv_pk = ws + 2 * 512 * 512;
  half_t* w1_pk = ws + 3 * 512 * 512;

  pack_b_kernel<<<512 * 512 / 256, 256, 0, stream>>>(Wq, wq_pk, 512);
  pack_b_kernel<<<512 * 512 / 256, 256, 0, stream>>>(Wk, wk_pk, 512);
  pack_b_kernel<<<512 * 512 / 256, 256, 0, stream>>>(Wv, wv_pk, 512);
  pack_b_kernel<<<512 * 256 / 256, 256, 0, stream>>>(W1, w1_pk, 256);

  (void)hipFuncSetAttribute((const void*)trust_kernel,
                            hipFuncAttributeMaxDynamicSharedMemorySize, SMEM_BYTES);
  trust_kernel<<<E_TOT / ETILE, 256, SMEM_BYTES, stream>>>(
      h_i, evo, var_i, var_j, bq, bk, bv, logd, b1, W2, b2,
      wq_pk, wk_pk, wv_pk, w1_pk, out);
}
